// TrendEncoder_79310866087965
// MI455X (gfx1250) — compile-verified, matrix operand order checked by reading, not run on hardware
//
#include <hip/hip_runtime.h>
#include <hip/hip_bf16.h>
#include <math.h>

typedef __attribute__((ext_vector_type(16))) _Float16 v16h;
typedef __attribute__((ext_vector_type(8)))  _Float16 v8h;
typedef __attribute__((ext_vector_type(8)))  float    v8f;

#define BATCH   4096
#define SEQ     200
#define BINS    256
#define HID     64
#define STEPF   4.0f
#define ROWS    16      // batch rows per block (one WMMA M-tile)
#define NTHR    256     // 8 waves of 32
#define HSTR    72      // padded h stride in halves (bank spread, 16B aligned)
#define GSTR    260     // padded gates/hist stride in floats (bank spread)

#define LOG2E   1.4426950408889634f

// sigmoid(x) = 1 / (1 + 2^(-x*log2e))   -> v_exp_f32 + v_rcp_f32, branch-free
__device__ __forceinline__ float fast_sigmoid(float x) {
    float e = __builtin_amdgcn_exp2f(-x * LOG2E);
    return __builtin_amdgcn_rcpf(1.0f + e);
}

// tanh: native v_tanh_f32 on gfx1250 if the builtin exists, else exp2/rcp form
__device__ __forceinline__ float fast_tanh(float x) {
#if __has_builtin(__builtin_amdgcn_tanhf)
    return __builtin_amdgcn_tanhf(x);
#else
    // tanh(x) = 1 - 2/(e^{2x}+1) = 1 - 2*rcp(2^(2x*log2e)+1)
    float e = __builtin_amdgcn_exp2f(x * (2.0f * LOG2E));
    return 1.0f - 2.0f * __builtin_amdgcn_rcpf(e + 1.0f);
#endif
}

__global__ __launch_bounds__(NTHR)
void trend_lstm_fused(const float* __restrict__ time_,
                      const int*   __restrict__ length,
                      const float* __restrict__ predict_time,
                      const float* __restrict__ W_ih,
                      const float* __restrict__ W_hh,
                      const float* __restrict__ b_ih,
                      const float* __restrict__ b_hh,
                      float* __restrict__ out)
{
    __shared__ float    hist [ROWS * GSTR];   // per-row time histogram (LSTM input seq)
    __shared__ float    gates[ROWS * GSTR];   // [16 rows][256 gate outputs] (padded)
    __shared__ _Float16 hsh  [ROWS * HSTR];   // hidden state, f16, padded rows

    const int tid  = threadIdx.x;
    const int row0 = blockIdx.x * ROWS;

    // ---- zero LDS state ----
    for (int i = tid; i < ROWS * GSTR; i += NTHR) hist[i] = 0.0f;
    for (int i = tid; i < ROWS * HSTR; i += NTHR) hsh[i]  = (_Float16)0.0f;
    __syncthreads();

    // ---- fused histogram build (LDS float atomics -> ds_add_f32) ----
    for (int e = tid; e < ROWS * SEQ; e += NTHR) {
        int r  = e / SEQ;
        int j  = e - r * SEQ;
        int rg = row0 + r;
        if (j < length[rg]) {
            float pt  = predict_time[rg];
            float tv  = time_[rg * SEQ + j];
            int   pos = (int)((pt - tv) / STEPF);   // trunc toward 0, like Python int()
            if (pos >= 0 && pos < BINS)
                atomicAdd(&hist[r * GSTR + (BINS - 1 - pos)], 1.0f);
        }
    }

    // ---- preload W_hh B-fragments into registers (once, reused 256 steps) ----
    const int lane = tid & 31;
    const int wave = tid >> 5;     // 0..7 : owns gate columns [wave*32, wave*32+32)
    const int nrow = lane & 15;    // M (for A/D) or N (for B) index within tile
    const int hg   = lane >> 4;    // half-group 0/1

    v16h bfrag[2][2];              // [ntile][kfrag]
    #pragma unroll
    for (int nt = 0; nt < 2; ++nt) {
        int ncol = wave * 32 + nt * 16 + nrow;        // gate-output index = W_hh row
        const float* wr = W_hh + ncol * HID;
        #pragma unroll
        for (int kf = 0; kf < 2; ++kf) {
            int kb = kf * 32;
            v16h b;
            #pragma unroll
            for (int j = 0; j < 8; ++j) b[j]     = (_Float16)wr[kb + hg * 8 + j];
            #pragma unroll
            for (int j = 0; j < 8; ++j) b[8 + j] = (_Float16)wr[kb + 16 + hg * 8 + j];
            bfrag[nt][kf] = b;
        }
    }

    // ---- per-thread elementwise ownership: row = tid%16, units u0..u0+3 ----
    const int erow = tid & 15;
    const int u0   = (tid >> 4) * 4;
    float cst[4], bi[4], bf2[4], bg[4], bo[4], wii[4], wif[4], wig[4], wio[4];
    #pragma unroll
    for (int k = 0; k < 4; ++k) {
        int u   = u0 + k;
        cst[k]  = 0.0f;
        bi[k]   = b_ih[u]       + b_hh[u];
        bf2[k]  = b_ih[u + 64]  + b_hh[u + 64];
        bg[k]   = b_ih[u + 128] + b_hh[u + 128];
        bo[k]   = b_ih[u + 192] + b_hh[u + 192];
        wii[k]  = W_ih[u];
        wif[k]  = W_ih[u + 64];
        wig[k]  = W_ih[u + 128];
        wio[k]  = W_ih[u + 192];
    }
    float hout[4] = {0.0f, 0.0f, 0.0f, 0.0f};
    __syncthreads();

    // ---- 256-step LSTM recurrence ----
    for (int t = 0; t < BINS; ++t) {
        // A fragments (16x32 f16) from shared h; two 16-byte LDS loads each
        v16h a[2];
        #pragma unroll
        for (int kf = 0; kf < 2; ++kf) {
            const _Float16* hp = &hsh[nrow * HSTR + kf * 32 + hg * 8];
            v8h lo = *(const v8h*)(hp);
            v8h hi = *(const v8h*)(hp + 16);
            a[kf]  = __builtin_shufflevector(lo, hi,
                       0,1,2,3,4,5,6,7,8,9,10,11,12,13,14,15);
        }

        // gates[16 x 32-col slice] = h @ W_hh^T, K=64 chained over two WMMAs
        #pragma unroll
        for (int nt = 0; nt < 2; ++nt) {
            v8f acc = {};
            acc = __builtin_amdgcn_wmma_f32_16x16x32_f16(
                      false, a[0], false, bfrag[nt][0], (short)0, acc, false, false);
            acc = __builtin_amdgcn_wmma_f32_16x16x32_f16(
                      false, a[1], false, bfrag[nt][1], (short)0, acc, false, false);
            int ng = wave * 32 + nt * 16 + nrow;       // global gate column
            #pragma unroll
            for (int v = 0; v < 8; ++v)                // D layout: M = v + 8*hg
                gates[(v + 8 * hg) * GSTR + ng] = acc[v];
        }
        __syncthreads();  // gates visible; h reads done before overwrite below

        // elementwise LSTM cell update (branch-free fast transcendentals)
        float x = hist[erow * GSTR + t];
        const float* gr = &gates[erow * GSTR];
        #pragma unroll
        for (int k = 0; k < 4; ++k) {
            int u = u0 + k;
            float gi = gr[u]        + x * wii[k] + bi[k];
            float gf = gr[u + 64]   + x * wif[k] + bf2[k];
            float gg = gr[u + 128]  + x * wig[k] + bg[k];
            float go = gr[u + 192]  + x * wio[k] + bo[k];
            float si = fast_sigmoid(gi);
            float sf = fast_sigmoid(gf);
            float so = fast_sigmoid(go);
            float c  = sf * cst[k] + si * fast_tanh(gg);
            cst[k]   = c;
            float h  = so * fast_tanh(c);
            hout[k]  = h;
            hsh[erow * HSTR + u] = (_Float16)h;
        }
        __syncthreads();  // h visible for next step; gate reads done before rewrite
    }

    // ---- final hidden state out[B, 64] ----
    #pragma unroll
    for (int k = 0; k < 4; ++k)
        out[(row0 + erow) * HID + u0 + k] = hout[k];
}

extern "C" void kernel_launch(void* const* d_in, const int* in_sizes, int n_in,
                              void* d_out, int out_size, void* d_ws, size_t ws_size,
                              hipStream_t stream) {
    (void)in_sizes; (void)n_in; (void)d_ws; (void)ws_size;
    const float* time_        = (const float*)d_in[0];
    const int*   length       = (const int*)  d_in[1];
    const float* predict_time = (const float*)d_in[2];
    const float* W_ih         = (const float*)d_in[3];
    const float* W_hh         = (const float*)d_in[4];
    const float* b_ih         = (const float*)d_in[5];
    const float* b_hh         = (const float*)d_in[6];
    float* out = (float*)d_out;
    (void)out_size;

    dim3 grid(BATCH / ROWS);   // 256 blocks, 16 batch rows each
    dim3 block(NTHR);          // 8 wave32 waves
    trend_lstm_fused<<<grid, block, 0, stream>>>(
        time_, length, predict_time, W_ih, W_hh, b_ih, b_hh, out);
}
